// TwoDimensionalSSM_2817498546814
// MI455X (gfx1250) — matrix operand order repaired and checked by reading, not compiled, not run on hardware
//
#include <hip/hip_runtime.h>
#include <hip/hip_fp16.h>
#include <math.h>

typedef __attribute__((ext_vector_type(16))) _Float16 v16h;
typedef __attribute__((ext_vector_type(8)))  _Float16 v8h;
typedef __attribute__((ext_vector_type(8)))  float    v8f;
typedef __attribute__((ext_vector_type(4)))  unsigned int v4u;
typedef __attribute__((ext_vector_type(8)))  int v8i;
typedef __attribute__((ext_vector_type(4)))  int v4i;

#define LSIDE 32
#define SEQ   1024
#define BATCH 16
#define EMBED 768
#define NDIMS 16
#define KTOT  3072
#define KSCALE 0.25f   /* sqrt(1/16) */
#define ACHUNK 32      /* full lag set resident: 64 KB Toeplitz tiles in LDS */

struct c32 { float x, y; };
__device__ __forceinline__ c32 cmul(c32 a, c32 b) { return { a.x*b.x - a.y*b.y, a.x*b.y + a.y*b.x }; }
__device__ __forceinline__ c32 cadd(c32 a, c32 b) { return { a.x + b.x, a.y + b.y }; }
__device__ __forceinline__ c32 shfl_c(c32 v, int src) {
    c32 r; r.x = __shfl(v.x, src, 32); r.y = __shfl(v.y, src, 32); return r;
}
__device__ __forceinline__ float sigmoidf(float v) { return 1.0f / (1.0f + expf(-v)); }

// ---------------------------------------------------------------------------
// K1: build SSM impulse-response kernel k[h][i][j], h = dir*768 + d.
// One block per h; wave n = state dim; lane = column j. Deterministic LDS
// reduction over the 16 state dims (no float atomics -> bitwise repeatable).
// ---------------------------------------------------------------------------
__global__ void __launch_bounds__(512)
ssm_kernel_build(const float* __restrict__ A_angle, const float* __restrict__ A_radius,
                 const float* __restrict__ B1p, const float* __restrict__ B2p,
                 const float* __restrict__ C1p, const float* __restrict__ C2p,
                 float* __restrict__ kout) {
    __shared__ float kred[NDIMS * LSIDE];   // per-(n,j) partials for one row i
    const int h    = blockIdx.x;
    const int tid  = threadIdx.x;
    const int n    = tid >> 5;
    const int lane = tid & 31;

    c32 A[4];
#pragma unroll
    for (int c = 0; c < 4; ++c) {
        float ang = sigmoidf(A_angle[(c * KTOT + h) * NDIMS + n]) * 6.283185307179586f;
        float rad = sigmoidf(A_radius[(c * KTOT + h) * NDIMS + n]);
        A[c] = { rad * cosf(ang), rad * sinf(ang) };
    }
    const int pb = (h * NDIMS + n) * 2;
    const c32 B1 = { sigmoidf(B1p[pb]), sigmoidf(B1p[pb + 1]) };
    const c32 B2 = { sigmoidf(B2p[pb]), sigmoidf(B2p[pb + 1]) };
    const c32 C1 = { C1p[pb], C1p[pb + 1] };
    const c32 C2 = { C2p[pb], C2p[pb + 1] };

    c32 xh_prev = {0.f, 0.f}, xv_prev = {0.f, 0.f};
    for (int i = 0; i < LSIDE; ++i) {
        // vertical flow: xv[j] = A3*xh_prev[j] + A4*xv_prev[j]  (+B2 impulse at 0,0)
        c32 xv = cadd(cmul(A[2], xh_prev), cmul(A[3], xv_prev));
        if (i == 0 && lane == 0) xv = cadd(xv, B2);
        // horizontal recurrence xh[j] = A1*xh[j-1] + c[j], c[j] = A2*xv[j-1] (j>0)
        c32 xvm1 = shfl_c(xv, lane - 1);
        c32 cj;
        if (lane == 0) cj = (i == 0) ? B1 : c32{0.f, 0.f};
        else           cj = cmul(A[1], xvm1);
        c32 s = cj, ap = A[0];
#pragma unroll
        for (int off = 1; off < 32; off <<= 1) {      // Hillis-Steele linear-recurrence scan
            c32 t = shfl_c(s, lane - off);
            if (lane >= off) s = cadd(s, cmul(ap, t));
            ap = cmul(ap, ap);
        }
        c32 xh = s;
        kred[n * LSIDE + lane] =
            KSCALE * (C1.x * xh.x - C1.y * xh.y + C2.x * xv.x - C2.y * xv.y);
        __syncthreads();
        if (tid < LSIDE) {                            // deterministic reduce over n
            float acc = 0.f;
#pragma unroll
            for (int nn = 0; nn < NDIMS; ++nn) acc += kred[nn * LSIDE + tid];
            float sc = ((i == 0) ? 2.f : 1.f) * ((tid == 0) ? 2.f : 1.f);
            if (i == 0 && tid == 0) sc *= 0.25f;      // boundary corrections
            kout[h * SEQ + i * LSIDE + tid] = acc * sc;
        }
        __syncthreads();
        xh_prev = xh; xv_prev = xv;
    }
}

// ---------------------------------------------------------------------------
// K2: transpose x (S,B,D) f32 -> XT (D,B,S) f16, coalesced both sides.
// ---------------------------------------------------------------------------
__global__ void __launch_bounds__(256)
transpose_x_kernel(const float* __restrict__ x, _Float16* __restrict__ XT) {
    __shared__ float tile[32][33];
    const int b  = blockIdx.z;
    const int d0 = blockIdx.x * 32;
    const int s0 = blockIdx.y * 32;
    for (int row = threadIdx.y; row < 32; row += 8)
        tile[row][threadIdx.x] = x[((size_t)(s0 + row) * BATCH + b) * EMBED + d0 + threadIdx.x];
    __syncthreads();
    for (int row = threadIdx.y; row < 32; row += 8)
        XT[(size_t)(d0 + row) * (BATCH * SEQ) + b * SEQ + s0 + threadIdx.x] =
            (_Float16)tile[threadIdx.x][row];
}

// ---------------------------------------------------------------------------
// K3: main conv. One WG per channel d (8 waves). For each direction, build
// triangular-Toeplitz tiles T(a) (or T^T) in LDS from k, then accumulate
//   O_i[j,b] += T(a) @ X_{i -/+ a}   via v_wmma_f32_16x16x32_f16.
// The x tile is staged into LDS by the Tensor Data Mover (one D#-described
// DMA issued by wave 0) instead of a per-thread copy loop.
// Epilogue fuses residual (x*omega) + SiLU; writes D-major (coalesced).
// ---------------------------------------------------------------------------
__global__ void __launch_bounds__(256)
ssm_conv_wmma(const _Float16* __restrict__ XT, const float* __restrict__ kker,
              const float* __restrict__ omega, float* __restrict__ OT) {
    __shared__ __align__(64) _Float16 Xlds[BATCH * SEQ];            // [b][s]    32 KB
    __shared__ __align__(64) _Float16 Tlds[ACHUNK * LSIDE * LSIDE]; // [a][j][c] 64 KB
    __shared__ float kstage[LSIDE * LSIDE];                         //            4 KB
    const int d    = blockIdx.x;
    const int tid  = threadIdx.x;
    const int wave = tid >> 5;
    const int lane = tid & 31;
    const int g    = lane >> 4;     // lane half: selects contiguous K range 16*g..16*g+15 (same in A and B)
    const int lm   = lane & 15;

#if __has_builtin(__builtin_amdgcn_tensor_load_to_lds) && __has_builtin(__builtin_amdgcn_s_wait_tensorcnt)
    if (wave == 0) {
        // Tensor DMA descriptor (D#), ISA 8.3/8.4: 1 descriptor, 2-byte elems,
        // 1-D tile of 16384 elements: LDS[Xlds] <= XT[d * 16384 ..].
        const unsigned long long ga =
            (unsigned long long)(uintptr_t)(XT + (size_t)d * (BATCH * SEQ));
        const unsigned ldsa = (unsigned)(uintptr_t)&Xlds[0];
        v4u g0;
        g0[0] = 1u;                                       // count=1, user mode
        g0[1] = ldsa;                                     // lds_addr
        g0[2] = (unsigned)(ga & 0xffffffffu);             // global_addr[31:0]
        g0[3] = (unsigned)((ga >> 32) & 0x01ffffffu) | 0x80000000u; // addr[56:32] | type=2
        v8i g1;
        g1[0] = (int)(1u << 16);                          // data_size=1 (2 bytes)
        g1[1] = (int)(16384u << 16);                      // tensor_dim0[15:0] << 16
        g1[2] = (int)(1u << 16);                          // dim0 hi=0 | tensor_dim1=1
        g1[3] = (int)(16384u << 16);                      // dim1 hi=0 | tile_dim0=16384
        g1[4] = 1;                                        // tile_dim1=1, tile_dim2=0
        g1[5] = 16384;                                    // tensor_dim0_stride lo
        g1[6] = 0;                                        // stride0 hi | stride1 lo
        g1[7] = 0;
        v4i g2 = {0, 0, 0, 0};
        v4i g3 = {0, 0, 0, 0};
#if defined(__clang_major__) && (__clang_major__ >= 23)
        v8i g4 = {0, 0, 0, 0, 0, 0, 0, 0};
        __builtin_amdgcn_tensor_load_to_lds(g0, g1, g2, g3, g4, 0);
#else
        __builtin_amdgcn_tensor_load_to_lds(g0, g1, g2, g3, 0);
#endif
        __builtin_amdgcn_s_wait_tensorcnt(0);
    }
    __syncthreads();
#else
    { // fallback: stage this channel's x tile in 128b chunks
        const uint4* src = (const uint4*)(XT + (size_t)d * (BATCH * SEQ));
        uint4* dst = (uint4*)Xlds;
        for (int idx = tid; idx < (BATCH * SEQ) / 8; idx += 256) dst[idx] = src[idx];
    }
#endif

    const v8f zero8 = {0.f, 0.f, 0.f, 0.f, 0.f, 0.f, 0.f, 0.f};
    v8f acc[4][2];
#pragma unroll
    for (int s = 0; s < 4; ++s) { acc[s][0] = zero8; acc[s][1] = zero8; }

    for (int dir = 0; dir < 4; ++dir) {
        __syncthreads();
        const float* kbase = kker + (size_t)(dir * EMBED + d) * SEQ;
        if (dir < 3) __builtin_prefetch(kker + (size_t)((dir + 1) * EMBED + d) * SEQ, 0, 0);
        for (int idx = tid; idx < SEQ; idx += 256) kstage[idx] = kbase[idx];
        const bool jflip = (dir >= 2);   // dirs 2,3: column flip  -> T^T
        const bool iflip = (dir & 1);    // dirs 1,3: row flip     -> r = i + a

        for (int a0 = 0; a0 < LSIDE; a0 += ACHUNK) {
            __syncthreads();             // kstage ready; prior chunk's WMMA reads done
            for (int idx = tid; idx < ACHUNK * LSIDE * LSIDE; idx += 256) {
                int a = a0 + (idx >> 10);
                int j = (idx >> 5) & 31;
                int c = idx & 31;
                int m = jflip ? (c - j) : (j - c);
                Tlds[idx] = (_Float16)((m >= 0) ? kstage[a * LSIDE + m] : 0.0f);
            }
            __syncthreads();
#pragma unroll
            for (int slot = 0; slot < 4; ++slot) {
                const int i    = wave + slot * 8;
                const int amax = iflip ? (31 - i) : i;
                const int ahi  = (a0 + ACHUNK - 1 < amax) ? (a0 + ACHUNK - 1) : amax;
                for (int a = a0; a <= ahi; ++a) {
                    const int r = iflip ? (i + a) : (i - a);
                    // B (32K x 16N): lane holds N=lm, 16 contiguous K at g*16
                    const v16h Bm = *(const v16h*)&Xlds[lm * SEQ + r * LSIDE + (g << 4)];
                    // A (16M x 32K): lane holds M=lm, 16 contiguous K at g*16
                    const int rowA0 = (a - a0) * SEQ + lm * LSIDE;
                    const int rowA1 = rowA0 + (LSIDE << 4);   // +16 rows for M-tile 1
                    v8h a0lo = *(const v8h*)&Tlds[rowA0 + (g << 4)];
                    v8h a0hi = *(const v8h*)&Tlds[rowA0 + 8 + (g << 4)];
                    v8h a1lo = *(const v8h*)&Tlds[rowA1 + (g << 4)];
                    v8h a1hi = *(const v8h*)&Tlds[rowA1 + 8 + (g << 4)];
                    v16h Am0 = __builtin_shufflevector(a0lo, a0hi, 0,1,2,3,4,5,6,7,8,9,10,11,12,13,14,15);
                    v16h Am1 = __builtin_shufflevector(a1lo, a1hi, 0,1,2,3,4,5,6,7,8,9,10,11,12,13,14,15);
                    acc[slot][0] = __builtin_amdgcn_wmma_f32_16x16x32_f16(
                        false, Am0, false, Bm, (short)0, acc[slot][0], false, false);
                    acc[slot][1] = __builtin_amdgcn_wmma_f32_16x16x32_f16(
                        false, Am1, false, Bm, (short)0, acc[slot][1], false, false);
                }
            }
        }
    }

    // epilogue: z = conv + x*omega; out = silu(z); OT[d][b][s], two b128 stores/lane
    const float om = omega[d];
    const int b = lm;
#pragma unroll
    for (int slot = 0; slot < 4; ++slot) {
        const int i = wave + slot * 8;
#pragma unroll
        for (int t = 0; t < 2; ++t) {
            const int jbase = t * 16 + (g << 3);   // D-matrix: M = v + 8*(lane/16)
            const int s0 = i * LSIDE + jbase;
            float4 o[2];
#pragma unroll
            for (int v = 0; v < 8; ++v) {
                const int s = s0 + v;
                float xval = (float)Xlds[b * SEQ + s];
                float z = acc[slot][t][v] + xval * om;
                ((float*)o)[v] = z * (1.0f / (1.0f + expf(-z)));
            }
            float4* dst = (float4*)(OT + (size_t)d * (BATCH * SEQ) + b * SEQ + s0);
            dst[0] = o[0];
            dst[1] = o[1];
        }
    }
}

// ---------------------------------------------------------------------------
// K4: transpose OT (D,B,S) f32 -> out (S,B,D) f32, coalesced both sides.
// ---------------------------------------------------------------------------
__global__ void __launch_bounds__(256)
transpose_out_kernel(const float* __restrict__ OT, float* __restrict__ out) {
    __shared__ float tile[32][33];
    const int b  = blockIdx.z;
    const int d0 = blockIdx.x * 32;
    const int s0 = blockIdx.y * 32;
    for (int row = threadIdx.y; row < 32; row += 8)
        tile[row][threadIdx.x] = OT[(size_t)(d0 + row) * (BATCH * SEQ) + b * SEQ + s0 + threadIdx.x];
    __syncthreads();
    for (int row = threadIdx.y; row < 32; row += 8)
        out[((size_t)(s0 + row) * BATCH + b) * EMBED + d0 + threadIdx.x] = tile[threadIdx.x][row];
}

extern "C" void kernel_launch(void* const* d_in, const int* in_sizes, int n_in,
                              void* d_out, int out_size, void* d_ws, size_t ws_size,
                              hipStream_t stream) {
    const float* x        = (const float*)d_in[0];
    const float* A_angle  = (const float*)d_in[1];
    const float* A_radius = (const float*)d_in[2];
    const float* B1p      = (const float*)d_in[3];
    const float* B2p      = (const float*)d_in[4];
    const float* C1p      = (const float*)d_in[5];
    const float* C2p      = (const float*)d_in[6];
    const float* omega    = (const float*)d_in[7];
    float* out = (float*)d_out;

    char* ws = (char*)d_ws;
    float*    kbuf = (float*)ws;                               // 3072*1024*4  = 12,582,912 B
    _Float16* XT   = (_Float16*)(ws + 12582912);               // 768*16384*2  = 25,165,824 B
    float*    OT   = (float*)(ws + 12582912 + 25165824);       // 768*16384*4  = 50,331,648 B

    ssm_kernel_build<<<dim3(KTOT), dim3(512), 0, stream>>>(A_angle, A_radius, B1p, B2p, C1p, C2p, kbuf);
    transpose_x_kernel<<<dim3(EMBED / 32, SEQ / 32, BATCH), dim3(32, 8), 0, stream>>>(x, XT);
    ssm_conv_wmma<<<dim3(EMBED), dim3(256), 0, stream>>>(XT, kbuf, omega, OT);
    transpose_out_kernel<<<dim3(EMBED / 32, SEQ / 32, BATCH), dim3(32, 8), 0, stream>>>(OT, out);
}
